// MultiHeadAttention_8710193676737
// MI455X (gfx1250) — compile-verified, matrix operand order checked by reading, not run on hardware
//
#include <hip/hip_runtime.h>
#include <hip/hip_bf16.h>

// Problem constants (match reference)
#define BB   4
#define LL   2048
#define DDIM 512
#define HH   8
#define DKV  64
#define NEGV (-1e9f)
// 1 / sqrt(2*DK) = 1/sqrt(128)
#define INV_TEMP 0.08838834764831845f

#ifndef __has_builtin
#define __has_builtin(x) 0
#endif

// CDNA5 async global->LDS path (ASYNCcnt). Guarded so both toolchains and the
// host pass compile; fallback is a synchronous LDS copy under the same barriers.
#if defined(__gfx1250__) && __has_builtin(__builtin_amdgcn_global_load_async_to_lds_b128)
#define HAVE_ASYNC_LDS 1
#else
#define HAVE_ASYNC_LDS 0
#endif

typedef __attribute__((ext_vector_type(16))) _Float16 v16h;
typedef __attribute__((ext_vector_type(8)))  float    v8f;
typedef __attribute__((ext_vector_type(4)))  int      v4i_a;

union FragAB {
    v16h     v;
    uint4    q[2];
    _Float16 h[16];
};
union FragC {
    v8f   v;
    float f[8];
};

__device__ __forceinline__ void zeroC(FragC& c) {
#pragma unroll
    for (int i = 0; i < 8; ++i) c.f[i] = 0.0f;
}

// D = A*B + C, 16x16x32 f16 -> f32 accumulate
__device__ __forceinline__ void wmma16(FragC& c, const FragAB& a, const FragAB& b) {
    c.v = __builtin_amdgcn_wmma_f32_16x16x32_f16(
        /*neg_a=*/false, a.v, /*neg_b=*/false, b.v,
        /*c_mod=*/(short)0, c.v, /*reuse_a=*/false, /*reuse_b=*/false);
}

// Load a 16x32 f16 fragment from row-major f16 matrix (row already includes lane&15).
// Per ISA: lane holds K-chunks [c0, c0+8) and [c0+16, c0+24), c0 = (lane>=16)*8.
__device__ __forceinline__ void frag_load_f16(FragAB& f, const _Float16* base, int ld,
                                              int row, int kk, int lane) {
    const int c0 = (lane & 16) ? 8 : 0;
    const _Float16* p = base + (size_t)row * ld + kk + c0;
    f.q[0] = *(const uint4*)(p);
    f.q[1] = *(const uint4*)(p + 16);
}

// Fragment from an LDS-staged 64x32 f16 tile (row stride 32 elements = 64 B).
__device__ __forceinline__ void frag_load_lds(FragAB& f, const _Float16* s, int row,
                                              int lane) {
    const int c0 = (lane & 16) ? 8 : 0;
    const _Float16* p = s + row * 32 + c0;
    f.q[0] = *(const uint4*)(p);
    f.q[1] = *(const uint4*)(p + 16);
}

// Same fragment, but source matrix is f32: convert in registers.
__device__ __forceinline__ void frag_load_f32(FragAB& f, const float* base, int ld,
                                              int row, int kk, int lane) {
    const int c0 = (lane & 16) ? 8 : 0;
    const float* p = base + (size_t)row * ld + kk + c0;
    float4 a0 = *(const float4*)(p);
    float4 a1 = *(const float4*)(p + 4);
    float4 b0 = *(const float4*)(p + 16);
    float4 b1 = *(const float4*)(p + 20);
    f.h[0]  = (_Float16)a0.x; f.h[1]  = (_Float16)a0.y;
    f.h[2]  = (_Float16)a0.z; f.h[3]  = (_Float16)a0.w;
    f.h[4]  = (_Float16)a1.x; f.h[5]  = (_Float16)a1.y;
    f.h[6]  = (_Float16)a1.z; f.h[7]  = (_Float16)a1.w;
    f.h[8]  = (_Float16)b0.x; f.h[9]  = (_Float16)b0.y;
    f.h[10] = (_Float16)b0.z; f.h[11] = (_Float16)b0.w;
    f.h[12] = (_Float16)b1.x; f.h[13] = (_Float16)b1.y;
    f.h[14] = (_Float16)b1.z; f.h[15] = (_Float16)b1.w;
}

// Wait for this wave's outstanding async global->LDS transfers.
__device__ __forceinline__ void wait_async_all() {
#if HAVE_ASYNC_LDS
#if __has_builtin(__builtin_amdgcn_s_wait_asynccnt)
    __builtin_amdgcn_s_wait_asynccnt(0);
#else
    asm volatile("s_wait_asynccnt 0x0" ::: "memory");
#endif
#endif
}

// Cooperative 64x32 f16 tile fetch: 256 threads x 16B. Async on CDNA5.
__device__ __forceinline__ void stage_tile(const _Float16* __restrict__ g, int kk,
                                           _Float16* s, int tid) {
    const int r = tid >> 2;          // 0..63
    const int c = (tid & 3) * 8;     // 0,8,16,24
    const _Float16* gp = g + (size_t)r * LL + kk + c;
    _Float16* sp = s + r * 32 + c;
#if HAVE_ASYNC_LDS
    // Toolchain signature: (int4 AS1*, int4 AS3*, imm offset, imm cpol)
    __builtin_amdgcn_global_load_async_to_lds_b128(
        (__attribute__((address_space(1))) v4i_a*)gp,
        (__attribute__((address_space(3))) v4i_a*)sp, 0, 0);
#else
    *(uint4*)sp = *(const uint4*)gp;
#endif
}

// ---------------------------------------------------------------------------
// Generic GEMM: C[M,N] = A[M,K] * B^T + bias (B given as N x K row-major).
// M = 8192, N = 512, K = 512. Block = 64x64 tile, 8 waves x 2 subtiles.
// AF32: A operand is f32 (convert on the fly) vs f16.
// MODE 0: out f16 head-split  [(b*H+h)*L + l]*64 + c   (Q/K layout)
// MODE 1: out f16 head-split transposed [(b*H+h)*64 + c]*L + l  (V^T layout)
// MODE 2: out f32 plain [m*512+n] with bias + residual (FC layout)
// ---------------------------------------------------------------------------
template <int AF32, int MODE>
__global__ __launch_bounds__(256) void gemm_kernel(const void* __restrict__ Aptr,
                                                   const float* __restrict__ Bw,
                                                   const float* __restrict__ bias,
                                                   const float* __restrict__ resid,
                                                   void* __restrict__ outp) {
    const int lane  = threadIdx.x & 31;
    const int w     = threadIdx.x >> 5;
    const int ntile = blockIdx.x * 64;
    const int mtile = blockIdx.y * 64;
    const int Mi    = w & 3;
    const int Ni0   = (w >> 2) * 2;
    const int l15   = lane & 15;
    const int arow  = mtile + Mi * 16 + l15;

    FragC acc[2];
    zeroC(acc[0]);
    zeroC(acc[1]);

    for (int kk = 0; kk < 512; kk += 32) {
        FragAB a;
        if (AF32) frag_load_f32(a, (const float*)Aptr, 512, arow, kk, lane);
        else      frag_load_f16(a, (const _Float16*)Aptr, 512, arow, kk, lane);
#pragma unroll
        for (int t = 0; t < 2; ++t) {
            FragAB b;
            frag_load_f32(b, Bw, 512, ntile + (Ni0 + t) * 16 + l15, kk, lane);
            wmma16(acc[t], a, b);
        }
    }

    const int hi8 = (lane & 16) ? 8 : 0;
#pragma unroll
    for (int t = 0; t < 2; ++t) {
        const int n = ntile + (Ni0 + t) * 16 + l15;
        const float bv = bias[n];
#pragma unroll
        for (int i = 0; i < 8; ++i) {
            const int m = mtile + Mi * 16 + hi8 + i;
            float val = acc[t].f[i] + bv;
            if (MODE == 2) {
                ((float*)outp)[(size_t)m * 512 + n] = val + resid[(size_t)m * 512 + n];
            } else {
                const int bI = m >> 11;   // m / L
                const int l  = m & (LL - 1);
                const int h  = n >> 6;
                const int c  = n & 63;
                const _Float16 hv = (_Float16)val;
                if (MODE == 0)
                    ((_Float16*)outp)[((size_t)(bI * HH + h) * LL + l) * 64 + c] = hv;
                else
                    ((_Float16*)outp)[((size_t)(bI * HH + h) * 64 + c) * LL + l] = hv;
            }
        }
    }
}

// ---------------------------------------------------------------------------
// Scores: S[b,h,q,k] = (QR.KR^T + QP.KP^T)/TEMP, masked; written to d_out attn
// region in [H,B,L,L] layout. Real+phase chained into one WMMA accumulator.
// grid = (L/64, L/64, B*H)
// ---------------------------------------------------------------------------
__global__ __launch_bounds__(256) void scores_kernel(const _Float16* __restrict__ qr,
                                                     const _Float16* __restrict__ kr,
                                                     const _Float16* __restrict__ qp,
                                                     const _Float16* __restrict__ kp,
                                                     const int* __restrict__ mask,
                                                     float* __restrict__ attn) {
    const int lane = threadIdx.x & 31;
    const int w    = threadIdx.x >> 5;
    const int kt   = blockIdx.x * 64;
    const int qt   = blockIdx.y * 64;
    const int z    = blockIdx.z;              // b*H + h
    const int b    = z >> 3;
    const int h    = z & 7;
    const size_t hb = (size_t)z * LL * 64;
    const int Mi = w & 3, Ni0 = (w >> 2) * 2, l15 = lane & 15;
    const int arow = qt + Mi * 16 + l15;

    FragC acc[2];
    zeroC(acc[0]);
    zeroC(acc[1]);

#pragma unroll
    for (int kk = 0; kk < DKV; kk += 32) {
        FragAB ar, ap;
        frag_load_f16(ar, qr + hb, 64, arow, kk, lane);
        frag_load_f16(ap, qp + hb, 64, arow, kk, lane);
#pragma unroll
        for (int t = 0; t < 2; ++t) {
            const int brow = kt + (Ni0 + t) * 16 + l15;
            FragAB br, bp;
            frag_load_f16(br, kr + hb, 64, brow, kk, lane);
            frag_load_f16(bp, kp + hb, 64, brow, kk, lane);
            wmma16(acc[t], ar, br);
            wmma16(acc[t], ap, bp);
        }
    }

    const int hi8 = (lane & 16) ? 8 : 0;
    float* abase = attn + (size_t)(h * BB + b) * LL * LL;
#pragma unroll
    for (int t = 0; t < 2; ++t) {
        const int k = kt + (Ni0 + t) * 16 + l15;
#pragma unroll
        for (int i = 0; i < 8; ++i) {
            const int q = qt + Mi * 16 + hi8 + i;
            const float s = acc[t].f[i] * INV_TEMP;
            const int mv = mask[((size_t)b * LL + q) * LL + k];
            abase[(size_t)q * LL + k] = mv ? NEGV : s;
        }
    }
}

// ---------------------------------------------------------------------------
// Row softmax over 2048 elements (8 per thread in registers, single R+W pass).
// grid = H*B*L blocks.
// ---------------------------------------------------------------------------
__global__ __launch_bounds__(256) void softmax_kernel(float* __restrict__ attn) {
    float* p = attn + (size_t)blockIdx.x * LL;
    const int t = threadIdx.x;
    __shared__ float red[256];

    float x[8];
    float mx = -3.4e38f;
#pragma unroll
    for (int j = 0; j < 8; ++j) {
        x[j] = p[t + 256 * j];
        mx = fmaxf(mx, x[j]);
    }
    red[t] = mx;
    __syncthreads();
    for (int s = 128; s > 0; s >>= 1) {
        if (t < s) red[t] = fmaxf(red[t], red[t + s]);
        __syncthreads();
    }
    mx = red[0];
    __syncthreads();

    float sum = 0.0f;
#pragma unroll
    for (int j = 0; j < 8; ++j) {
        x[j] = __expf(x[j] - mx);
        sum += x[j];
    }
    red[t] = sum;
    __syncthreads();
    for (int s = 128; s > 0; s >>= 1) {
        if (t < s) red[t] += red[t + s];
        __syncthreads();
    }
    const float inv = 1.0f / red[0];
#pragma unroll
    for (int j = 0; j < 8; ++j) p[t + 256 * j] = x[j] * inv;
}

// ---------------------------------------------------------------------------
// attn @ V for real and phase simultaneously (shared f32->f16 A fragments).
// V^T tiles (64x32, real+phase) are double-buffered in LDS via CDNA5 async
// global->LDS loads (ASYNCcnt), removing the 4x redundant global fetch of B
// fragments and overlapping tile fetch with WMMA.
// grid = (L/64, B*H)
// ---------------------------------------------------------------------------
__global__ __launch_bounds__(256) void attnv_kernel(const float* __restrict__ attn,
                                                    const _Float16* __restrict__ vrT,
                                                    const _Float16* __restrict__ vpT,
                                                    _Float16* __restrict__ cr,
                                                    _Float16* __restrict__ cp) {
    const int tid  = threadIdx.x;
    const int lane = tid & 31;
    const int w    = tid >> 5;
    const int qt   = blockIdx.x * 64;
    const int z    = blockIdx.y;              // b*H + h
    const int b    = z >> 3;
    const int h    = z & 7;
    const int Mi = w & 3, Ni0 = (w >> 2) * 2, l15 = lane & 15;

    const float*    A  = attn + (size_t)(h * BB + b) * LL * LL;  // [L x L]
    const _Float16* BR = vrT + (size_t)z * 64 * LL;              // [64 x L]
    const _Float16* BP = vpT + (size_t)z * 64 * LL;

    __shared__ __align__(16) _Float16 VsR[2][64 * 32];
    __shared__ __align__(16) _Float16 VsP[2][64 * 32];

    FragC accR[2], accP[2];
    zeroC(accR[0]); zeroC(accR[1]);
    zeroC(accP[0]); zeroC(accP[1]);

    // Prologue: stage first K-tile into buffer 0.
    stage_tile(BR, 0, VsR[0], tid);
    stage_tile(BP, 0, VsP[0], tid);

    const int arow = qt + Mi * 16 + l15;
    int buf = 0;
    for (int kk = 0; kk < LL; kk += 32, buf ^= 1) {
        wait_async_all();        // my async writes for VsX[buf] are done
        __syncthreads();         // everyone's writes visible

        if (kk + 32 < LL) {      // prefetch next K-tile into the other buffer
            stage_tile(BR, kk + 32, VsR[buf ^ 1], tid);
            stage_tile(BP, kk + 32, VsP[buf ^ 1], tid);
        }
        if (kk + 128 < LL)
            __builtin_prefetch(&A[(size_t)arow * LL + kk + 128], 0, 1);

        FragAB a;
        frag_load_f32(a, A, LL, arow, kk, lane);
#pragma unroll
        for (int t = 0; t < 2; ++t) {
            const int brow = (Ni0 + t) * 16 + l15;
            FragAB br, bp;
            frag_load_lds(br, VsR[buf], brow, lane);
            frag_load_lds(bp, VsP[buf], brow, lane);
            wmma16(accR[t], a, br);
            wmma16(accP[t], a, bp);
        }
        __syncthreads();         // readers done before buf is rewritten
    }

    const int hi8 = (lane & 16) ? 8 : 0;
#pragma unroll
    for (int t = 0; t < 2; ++t) {
        const int d = (Ni0 + t) * 16 + l15;
#pragma unroll
        for (int i = 0; i < 8; ++i) {
            const int q = qt + Mi * 16 + hi8 + i;
            const size_t o = ((size_t)b * LL + q) * (HH * DKV) + h * 64 + d;
            cr[o] = (_Float16)accR[t].f[i];
            cp[o] = (_Float16)accP[t].f[i];
        }
    }
}

// ---------------------------------------------------------------------------
// LayerNorm over last dim (512), 2 elements per thread. grid = B*L.
// ---------------------------------------------------------------------------
__global__ __launch_bounds__(256) void ln_kernel(const float* __restrict__ in,
                                                 const float* __restrict__ g,
                                                 const float* __restrict__ bta,
                                                 float* __restrict__ out) {
    const size_t m = blockIdx.x;
    const float* p = in + m * DDIM;
    const int t = threadIdx.x;
    __shared__ float red[256];

    const float a = p[t];
    const float c = p[t + 256];
    red[t] = a + c;
    __syncthreads();
    for (int s = 128; s > 0; s >>= 1) {
        if (t < s) red[t] += red[t + s];
        __syncthreads();
    }
    const float mu = red[0] * (1.0f / 512.0f);
    __syncthreads();

    const float d0 = a - mu, d1 = c - mu;
    red[t] = d0 * d0 + d1 * d1;
    __syncthreads();
    for (int s = 128; s > 0; s >>= 1) {
        if (t < s) red[t] += red[t + s];
        __syncthreads();
    }
    const float rstd = rsqrtf(red[0] * (1.0f / 512.0f) + 1e-5f);

    float* o = out + m * DDIM;
    o[t]       = g[t]       * d0 * rstd + bta[t];
    o[t + 256] = g[t + 256] * d1 * rstd + bta[t + 256];
}

// ---------------------------------------------------------------------------
extern "C" void kernel_launch(void* const* d_in, const int* in_sizes, int n_in,
                              void* d_out, int out_size, void* d_ws, size_t ws_size,
                              hipStream_t stream) {
    const float* q_real  = (const float*)d_in[0];
    const float* k_real  = (const float*)d_in[1];
    const float* v_real  = (const float*)d_in[2];
    const float* q_phase = (const float*)d_in[3];
    const float* k_phase = (const float*)d_in[4];
    const float* v_phase = (const float*)d_in[5];
    const int*   mask    = (const int*)d_in[6];
    const float* w_q  = (const float*)d_in[7];
    const float* b_q  = (const float*)d_in[8];
    const float* w_k  = (const float*)d_in[9];
    const float* b_k  = (const float*)d_in[10];
    const float* w_v  = (const float*)d_in[11];
    const float* b_v  = (const float*)d_in[12];
    const float* w_fc = (const float*)d_in[13];
    const float* b_fc = (const float*)d_in[14];
    const float* ln_g = (const float*)d_in[15];
    const float* ln_b = (const float*)d_in[16];

    float* out_r = (float*)d_out;
    float* out_p = out_r + (size_t)BB * LL * DDIM;
    float* attn  = out_p + (size_t)BB * LL * DDIM;   // [H, B, L, L]

    // Workspace carve-out (256B aligned sub-buffers, ~101 MB total)
    char*  ws  = (char*)d_ws;
    size_t off = 0;
    auto carve = [&](size_t bytes) -> void* {
        void* p = ws + off;
        off += (bytes + 255) & ~(size_t)255;
        return p;
    };
    const size_t projBytes = (size_t)BB * HH * LL * DKV * sizeof(_Float16); // 8.4 MB
    _Float16* qr  = (_Float16*)carve(projBytes);
    _Float16* kr  = (_Float16*)carve(projBytes);
    _Float16* qp  = (_Float16*)carve(projBytes);
    _Float16* kp  = (_Float16*)carve(projBytes);
    _Float16* vrT = (_Float16*)carve(projBytes);
    _Float16* vpT = (_Float16*)carve(projBytes);
    _Float16* cr  = (_Float16*)carve(projBytes);
    _Float16* cp  = (_Float16*)carve(projBytes);
    float*    fr  = (float*)carve((size_t)BB * LL * DDIM * sizeof(float));  // 16.8 MB
    float*    fp  = (float*)carve((size_t)BB * LL * DDIM * sizeof(float));

    const dim3 blk(256);
    const dim3 gg(DDIM / 64, (BB * LL) / 64);   // (8, 128) GEMM grid

    // 1) projections (A = f32 inputs, B = f32 weights, f16 WMMA on the fly)
    gemm_kernel<1, 0><<<gg, blk, 0, stream>>>(q_real,  w_q, b_q, nullptr, qr);
    gemm_kernel<1, 0><<<gg, blk, 0, stream>>>(k_real,  w_k, b_k, nullptr, kr);
    gemm_kernel<1, 1><<<gg, blk, 0, stream>>>(v_real,  w_v, b_v, nullptr, vrT);
    gemm_kernel<1, 0><<<gg, blk, 0, stream>>>(q_phase, w_q, b_q, nullptr, qp);
    gemm_kernel<1, 0><<<gg, blk, 0, stream>>>(k_phase, w_k, b_k, nullptr, kp);
    gemm_kernel<1, 1><<<gg, blk, 0, stream>>>(v_phase, w_v, b_v, nullptr, vpT);

    // 2) masked scores -> attn region (unnormalized)
    scores_kernel<<<dim3(LL / 64, LL / 64, BB * HH), blk, 0, stream>>>(qr, kr, qp, kp,
                                                                       mask, attn);
    // 3) row softmax in place
    softmax_kernel<<<dim3(HH * BB * LL), blk, 0, stream>>>(attn);

    // 4) attn @ V (real + phase) -> concat layout
    attnv_kernel<<<dim3(LL / 64, BB * HH), blk, 0, stream>>>(attn, vrT, vpT, cr, cp);

    // 5) FC + bias + residual (f32 out)
    gemm_kernel<0, 2><<<gg, blk, 0, stream>>>(cr, w_fc, b_fc, q_real,  fr);
    gemm_kernel<0, 2><<<gg, blk, 0, stream>>>(cp, w_fc, b_fc, q_phase, fp);

    // 6) LayerNorm -> final outputs
    ln_kernel<<<dim3(BB * LL), blk, 0, stream>>>(fr, ln_g, ln_b, out_r);
    ln_kernel<<<dim3(BB * LL), blk, 0, stream>>>(fp, ln_g, ln_b, out_p);
}
